// LSTMRewardModel_62285615727292
// MI455X (gfx1250) — compile-verified, matrix operand order checked by reading, not run on hardware
//
#include <hip/hip_runtime.h>
#include <hip/hip_bf16.h>

// ---------------------------------------------------------------------------
// CDNA5 (gfx1250) LSTM reward model.
//   Setup : pre-swizzle all MLP weights into bf16 WMMA-B fragment layout (ws)
//   Phase A: fused pre-MLP (37->256->128->128) + xWi = x3@Wi + bl  [token-par]
//            xWi stored in LSTM-fragment layout [bTile][t][col][row&15]
//   Phase B: LSTM recurrence over T, 16 batch rows per workgroup   [serial T]
//            Wh in LDS (bf16 frags), double-buffered h tile (1 barrier/step),
//            xWi double-buffered as float4 loads
//   Phase C: fused post-MLP (concat 256 ->128->64->64->1)          [token-par]
// All GEMMs via v_wmma_f32_16x16x32_bf16 (f32 accumulate).
// ---------------------------------------------------------------------------

typedef __attribute__((ext_vector_type(16))) __bf16 bf16x16;
typedef __attribute__((ext_vector_type(8)))  __bf16 bf16x8;
typedef __attribute__((ext_vector_type(8)))  float  f32x8;

#define WMMA_BF16(A, B, C) \
  __builtin_amdgcn_wmma_f32_16x16x32_bf16(false, (A), false, (B), (short)0, (C), false, false)

// ---------------- fragment helpers (wave32) --------------------------------
// A (16x32 bf16, row-major source): lane l -> row = l&15, K base = 8*(l>>4);
//   v[0..7] = K..K+7, v[8..15] = K+16..K+23.
__device__ __forceinline__ bf16x16 load_afrag(const __bf16* base, int ld, int row0, int kbase) {
  int l = threadIdx.x & 31;
  const __bf16* p = base + (size_t)(row0 + (l & 15)) * ld + kbase + ((l >> 4) << 3);
  bf16x8 lo = *(const bf16x8*)(p);
  bf16x8 hi = *(const bf16x8*)(p + 16);
  return __builtin_shufflevector(lo, hi, 0,1,2,3,4,5,6,7,8,9,10,11,12,13,14,15);
}

// B (32x16 bf16) from global f32 weight W[K][M] (slow path; setup/Wh only)
__device__ __forceinline__ bf16x16 load_bfrag_g(const float* __restrict__ W, int M,
                                                int kbase, int col0, int kmax) {
  int l = threadIdx.x & 31;
  int col = col0 + (l & 15);
  int k0 = kbase + ((l >> 4) << 3);
  bf16x16 v;
#pragma unroll
  for (int j = 0; j < 8; ++j) {
    int k = k0 + ((j < 4) ? (2 * j) : (8 + 2 * j));
    v[2 * j]     = (k     < kmax) ? (__bf16)W[(size_t)k       * M + col] : (__bf16)0.0f;
    v[2 * j + 1] = (k + 1 < kmax) ? (__bf16)W[(size_t)(k + 1) * M + col] : (__bf16)0.0f;
  }
  return v;
}

// B fragment from pre-swizzled bf16 fragment buffer: 2x b128 per lane.
__device__ __forceinline__ bf16x16 load_bfrag_w(const __bf16* __restrict__ Wf, int nkt,
                                                int ct, int kt) {
  int lane = threadIdx.x & 31;
  const __bf16* p = Wf + ((size_t)(ct * nkt + kt) * 32 + lane) * 16;
  bf16x8 lo = *(const bf16x8*)(p);
  bf16x8 hi = *(const bf16x8*)(p + 8);
  return __builtin_shufflevector(lo, hi, 0,1,2,3,4,5,6,7,8,9,10,11,12,13,14,15);
}

__device__ __forceinline__ void store16(__bf16* p, bf16x16 v) {
  bf16x8 lo = __builtin_shufflevector(v, v, 0,1,2,3,4,5,6,7);
  bf16x8 hi = __builtin_shufflevector(v, v, 8,9,10,11,12,13,14,15);
  *(bf16x8*)(p)     = lo;
  *(bf16x8*)(p + 8) = hi;
}

__device__ __forceinline__ f32x8 splat8(float x) {
  f32x8 v;
#pragma unroll
  for (int i = 0; i < 8; ++i) v[i] = x;
  return v;
}

// C/D (16x16 f32): VGPR r -> row = row0 + 8*(l>>4) + r, col = col0 + (l&15)
__device__ __forceinline__ void store_c_lds_bf16(__bf16* base, int ld, int row0, int col0,
                                                 f32x8 acc, bool relu) {
  int l = threadIdx.x & 31;
  int col = col0 + (l & 15);
  int rb = row0 + ((l >> 4) << 3);
#pragma unroll
  for (int r = 0; r < 8; ++r) {
    float x = acc[r];
    if (relu) x = fmaxf(x, 0.0f);
    base[(size_t)(rb + r) * ld + col] = (__bf16)x;
  }
}

__device__ __forceinline__ void store_c_lds_f32(float* base, int ld, int row0, int col0,
                                                f32x8 acc, bool relu) {
  int l = threadIdx.x & 31;
  int col = col0 + (l & 15);
  int rb = row0 + ((l >> 4) << 3);
#pragma unroll
  for (int r = 0; r < 8; ++r) {
    float x = acc[r];
    if (relu) x = fmaxf(x, 0.0f);
    base[(size_t)(rb + r) * ld + col] = x;
  }
}

__device__ __forceinline__ void store_c_glb_f32(float* __restrict__ base, int ld, size_t row0,
                                                int col0, f32x8 acc, bool relu) {
  int l = threadIdx.x & 31;
  int col = col0 + (l & 15);
  size_t rb = row0 + (size_t)((l >> 4) << 3);
#pragma unroll
  for (int r = 0; r < 8; ++r) {
    float x = acc[r];
    if (relu) x = fmaxf(x, 0.0f);
    base[(rb + r) * (size_t)ld + col] = x;
  }
}

// xWi store in LSTM fragment layout: idx = ((bTile*T + t)*512 + col)*16 + (b&15)
__device__ __forceinline__ void store_c_xwi(float* __restrict__ dst, size_t tokBase,
                                            int col0, f32x8 acc, int T) {
  int l = threadIdx.x & 31;
  int col = col0 + (l & 15);
  size_t tokb = tokBase + (size_t)((l >> 4) << 3);
#pragma unroll
  for (int r = 0; r < 8; ++r) {
    unsigned tok = (unsigned)(tokb + r);
    unsigned b = tok / (unsigned)T;
    unsigned t = tok - b * (unsigned)T;
    size_t idx = (((size_t)(b >> 4) * T + t) * 512 + col) * 16 + (b & 15);
    dst[idx] = acc[r];
  }
}

__device__ __forceinline__ float fast_sigmoid(float x) {
  return __builtin_amdgcn_rcpf(1.0f + __expf(-x));
}
__device__ __forceinline__ float fast_tanh(float x) {
  return 2.0f * __builtin_amdgcn_rcpf(1.0f + __expf(-2.0f * x)) - 1.0f;
}

// ===========================================================================
// Setup: swizzle one weight matrix into bf16 B-fragment layout.
// grid = (nct, nkt), block = 32 (one wave builds one 32x16 fragment).
// ===========================================================================
__global__ __launch_bounds__(32) void swizzle_w_kernel(const float* __restrict__ W,
                                                       __bf16* __restrict__ dst,
                                                       int M, int nkt, int kmax) {
  int ct = blockIdx.x, kt = blockIdx.y;
  bf16x16 v = load_bfrag_g(W, M, kt * 32, ct * 16, kmax);
  store16(dst + ((size_t)(ct * nkt + kt) * 32 + (threadIdx.x & 31)) * 16, v);
}

// ===========================================================================
// Phase A: pre-MLP + xWi.  64 tokens / block, 8 waves.
// LDS: in[64][64] bf16 | x1[64][256] bf16 | x2[64][128] bf16 | x3[64][128] bf16
// ===========================================================================
__global__ __launch_bounds__(256) void premlp_kernel(
    const float* __restrict__ states, const float* __restrict__ actions,
    const __bf16* __restrict__ W1f, const float* __restrict__ b1,
    const __bf16* __restrict__ W2f, const float* __restrict__ b2,
    const __bf16* __restrict__ W3f, const float* __restrict__ b3,
    const __bf16* __restrict__ Wif, const float* __restrict__ bl,
    float* __restrict__ x3g, float* __restrict__ xwig, int T) {
  extern __shared__ __align__(16) char smem[];
  __bf16* sIn = (__bf16*)smem;                 // 64*64
  __bf16* sX1 = sIn + 64 * 64;                 // 64*256
  __bf16* sX2 = sX1 + 64 * 256;                // 64*128
  __bf16* sX3 = sX2 + 64 * 128;                // 64*128

  const int tid  = threadIdx.x;
  const int wave = tid >> 5;
  const int lane = tid & 31;
  const size_t tok0 = (size_t)blockIdx.x * 64;

  // stage concat([states,actions]) zero-padded to K=64, converted to bf16
  for (int i = tid; i < 64 * 64; i += 256) {
    int row = i >> 6, k = i & 63;
    size_t tok = tok0 + row;
    float v = 0.0f;
    if (k < 29)      v = states[tok * 29 + k];
    else if (k < 37) v = actions[tok * 8 + (k - 29)];
    sIn[i] = (__bf16)v;
  }
  __syncthreads();

  // L1: [64x64] @ [37(->64) x 256]  (relu) -> sX1
#pragma unroll
  for (int cti = 0; cti < 2; ++cti) {
    int ct = wave + cti * 8;
    int col0 = ct * 16;
    bf16x16 B0 = load_bfrag_w(W1f, 2, ct, 0);
    bf16x16 B1 = load_bfrag_w(W1f, 2, ct, 1);
    float bias = b1[col0 + (lane & 15)];
#pragma unroll
    for (int rt = 0; rt < 4; ++rt) {
      f32x8 acc = splat8(bias);
      acc = WMMA_BF16(load_afrag(sIn, 64, rt * 16, 0), B0, acc);
      acc = WMMA_BF16(load_afrag(sIn, 64, rt * 16, 32), B1, acc);
      store_c_lds_bf16(sX1, 256, rt * 16, col0, acc, true);
    }
  }
  __syncthreads();

  // L2: [64x256] @ [256x128] (relu) -> sX2
  {
    int col0 = wave * 16;
    bf16x16 Bf[8];
#pragma unroll
    for (int kt = 0; kt < 8; ++kt) Bf[kt] = load_bfrag_w(W2f, 8, wave, kt);
    float bias = b2[col0 + (lane & 15)];
#pragma unroll
    for (int rt = 0; rt < 4; ++rt) {
      f32x8 acc = splat8(bias);
#pragma unroll
      for (int kt = 0; kt < 8; ++kt)
        acc = WMMA_BF16(load_afrag(sX1, 256, rt * 16, kt * 32), Bf[kt], acc);
      store_c_lds_bf16(sX2, 128, rt * 16, col0, acc, true);
    }
  }
  __syncthreads();

  // L3: [64x128] @ [128x128] (relu) -> sX3 (bf16) and x3g (f32)
  {
    int col0 = wave * 16;
    bf16x16 Bf[4];
#pragma unroll
    for (int kt = 0; kt < 4; ++kt) Bf[kt] = load_bfrag_w(W3f, 4, wave, kt);
    float bias = b3[col0 + (lane & 15)];
#pragma unroll
    for (int rt = 0; rt < 4; ++rt) {
      f32x8 acc = splat8(bias);
#pragma unroll
      for (int kt = 0; kt < 4; ++kt)
        acc = WMMA_BF16(load_afrag(sX2, 128, rt * 16, kt * 32), Bf[kt], acc);
      store_c_lds_bf16(sX3, 128, rt * 16, col0, acc, true);
      store_c_glb_f32(x3g, 128, tok0 + rt * 16, col0, acc, true);
    }
  }
  __syncthreads();

  // L_xWi: [64x128] @ [128x512] + bl (no relu) -> xwig in LSTM-fragment layout
#pragma unroll
  for (int cti = 0; cti < 4; ++cti) {
    int ct = wave + cti * 8;
    int col0 = ct * 16;
    bf16x16 Bf[4];
#pragma unroll
    for (int kt = 0; kt < 4; ++kt) Bf[kt] = load_bfrag_w(Wif, 4, ct, kt);
    float bias = bl[col0 + (lane & 15)];
#pragma unroll
    for (int rt = 0; rt < 4; ++rt) {
      f32x8 acc = splat8(bias);
#pragma unroll
      for (int kt = 0; kt < 4; ++kt)
        acc = WMMA_BF16(load_afrag(sX3, 128, rt * 16, kt * 32), Bf[kt], acc);
      store_c_xwi(xwig, tok0 + rt * 16, col0, acc, T);
    }
  }
}

// ===========================================================================
// Phase B: LSTM.  16 blocks x 256 threads; block handles batch rows [16b,16b+16).
// Wh pre-swizzled to WMMA-B fragment layout in LDS (bf16, 128KB); double-
// buffered 16x128 h tile -> ONE barrier per step.  xWi read as 2x float4 per
// gate per lane, double-buffered (prefetch t+1).
// ===========================================================================
__device__ __forceinline__ void lstm_step(
    int t, int T, const __bf16* __restrict__ sWh,
    const __bf16* __restrict__ sRead, __bf16* __restrict__ sWrite,
    const float* __restrict__ xwig, size_t stepBase,
    int wave, int lane, int rowb, int rowLoc, int colh, int laneOff,
    f32x8& c, float (&xw)[4][8], float* __restrict__ lstm_out) {
  // A fragments of h (all 128 K) from the read buffer
  bf16x16 A[4];
#pragma unroll
  for (int kt = 0; kt < 4; ++kt) A[kt] = load_afrag(sRead, 128, 0, kt * 32);

  // prefetch next step's xWi (2x b128 per gate)
  float xwn[4][8];
  if (t + 1 < T) {
    size_t nb = stepBase + 512 * 16;
#pragma unroll
    for (int g = 0; g < 4; ++g) {
      const float4 a = *(const float4*)(xwig + nb + (size_t)g * (128 * 16) + laneOff);
      const float4 b = *(const float4*)(xwig + nb + (size_t)g * (128 * 16) + laneOff + 4);
      xwn[g][0] = a.x; xwn[g][1] = a.y; xwn[g][2] = a.z; xwn[g][3] = a.w;
      xwn[g][4] = b.x; xwn[g][5] = b.y; xwn[g][6] = b.z; xwn[g][7] = b.w;
    }
  }

  // gates = xWi + h @ Wh
  f32x8 acc[4];
#pragma unroll
  for (int g = 0; g < 4; ++g) {
#pragma unroll
    for (int r = 0; r < 8; ++r) acc[g][r] = xw[g][r];
#pragma unroll
    for (int kt = 0; kt < 4; ++kt) {
      int f = wave * 16 + g * 4 + kt;
      const __bf16* p = sWh + ((size_t)f * 32 + lane) * 16;
      bf16x8 lo = *(const bf16x8*)(p);
      bf16x8 hi = *(const bf16x8*)(p + 8);
      bf16x16 B = __builtin_shufflevector(lo, hi, 0,1,2,3,4,5,6,7,8,9,10,11,12,13,14,15);
      acc[g] = WMMA_BF16(A[kt], B, acc[g]);
    }
  }

  // elementwise LSTM cell; write h to d_out and to the write buffer
#pragma unroll
  for (int r = 0; r < 8; ++r) {
    float si = fast_sigmoid(acc[0][r]);
    float sf = fast_sigmoid(acc[1][r]);
    float tg = fast_tanh(acc[2][r]);
    float so = fast_sigmoid(acc[3][r]);
    float cn = sf * c[r] + si * tg;
    c[r] = cn;
    float h = so * fast_tanh(cn);
    lstm_out[((size_t)(rowb + r) * T + t) * 128 + colh] = h;
    sWrite[(rowLoc + r) * 128 + colh] = (__bf16)h;
  }
#pragma unroll
  for (int g = 0; g < 4; ++g)
#pragma unroll
    for (int r = 0; r < 8; ++r) xw[g][r] = xwn[g][r];

  __syncthreads();   // h writes visible; next step reads the other buffer
}

__global__ __launch_bounds__(256) void lstm_kernel(
    const float* __restrict__ Wh, const float* __restrict__ xwig,
    float* __restrict__ lstm_out, int T) {
  extern __shared__ __align__(16) char smem[];
  __bf16* sWh = (__bf16*)smem;            // 128 frags * 32 lanes * 16 halves = 128KB
  __bf16* sHa = sWh + 128 * 512;          // 16 x 128 bf16 (buffer A)
  __bf16* sHb = sHa + 16 * 128;           // 16 x 128 bf16 (buffer B)

  const int lane  = threadIdx.x & 31;
  const int wave  = threadIdx.x >> 5;
  const int bTile = blockIdx.x;
  const int b0    = bTile * 16;

  // pre-swizzle Wh: wave w owns cols [128g + 16w, +16) for g=0..3
#pragma unroll
  for (int g = 0; g < 4; ++g)
#pragma unroll
    for (int kt = 0; kt < 4; ++kt) {
      bf16x16 v = load_bfrag_g(Wh, 512, kt * 32, g * 128 + wave * 16, 1 << 30);
      int f = wave * 16 + g * 4 + kt;
      store16(sWh + ((size_t)f * 32 + lane) * 16, v);
    }
  for (int i = threadIdx.x; i < 16 * 128; i += 256) sHa[i] = (__bf16)0.0f;
  __syncthreads();

  const int rowLoc  = (lane >> 4) << 3;        // 0 or 8
  const int rowb    = b0 + rowLoc;
  const int colh    = wave * 16 + (lane & 15); // this lane's hidden column
  const int laneOff = colh * 16 + rowLoc;      // within one step's 512*16 block
  f32x8 c = splat8(0.0f);

  size_t stepBase = (size_t)bTile * T * (512 * 16);
  const size_t stepStride = 512 * 16;

  float xw[4][8];
#pragma unroll
  for (int g = 0; g < 4; ++g) {
    const float4 a = *(const float4*)(xwig + stepBase + (size_t)g * (128 * 16) + laneOff);
    const float4 b = *(const float4*)(xwig + stepBase + (size_t)g * (128 * 16) + laneOff + 4);
    xw[g][0] = a.x; xw[g][1] = a.y; xw[g][2] = a.z; xw[g][3] = a.w;
    xw[g][4] = b.x; xw[g][5] = b.y; xw[g][6] = b.z; xw[g][7] = b.w;
  }

  for (int t = 0; t < T; t += 2) {
    lstm_step(t,     T, sWh, sHa, sHb, xwig, stepBase,
              wave, lane, rowb, rowLoc, colh, laneOff, c, xw, lstm_out);
    stepBase += stepStride;
    lstm_step(t + 1, T, sWh, sHb, sHa, xwig, stepBase,
              wave, lane, rowb, rowLoc, colh, laneOff, c, xw, lstm_out);
    stepBase += stepStride;
  }
}

// ===========================================================================
// Phase C: post-MLP.  64 tokens / block.
// LDS: cat[64][256] bf16 | y4[64][128] bf16 | y5[64][64] bf16 | y6[64][64] f32
// ===========================================================================
__global__ __launch_bounds__(256) void postmlp_kernel(
    const float* __restrict__ x3g, const float* __restrict__ hg,
    const __bf16* __restrict__ W4f, const float* __restrict__ b4,
    const __bf16* __restrict__ W5f, const float* __restrict__ b5,
    const __bf16* __restrict__ W6f, const float* __restrict__ b6,
    const float* __restrict__ Wo, const float* __restrict__ bo,
    float* __restrict__ outg) {
  extern __shared__ __align__(16) char smem[];
  __bf16* sCat = (__bf16*)smem;               // 64*256
  __bf16* sY4  = sCat + 64 * 256;             // 64*128
  __bf16* sY5  = sY4 + 64 * 128;              // 64*64
  float*  sY6  = (float*)(sY5 + 64 * 64);     // 64*64 f32

  const int tid  = threadIdx.x;
  const int wave = tid >> 5;
  const int lane = tid & 31;
  const size_t tok0 = (size_t)blockIdx.x * 64;

  for (int i = tid; i < 64 * 256; i += 256) {
    int row = i >> 8, k = i & 255;
    size_t tok = tok0 + row;
    float v = (k < 128) ? x3g[tok * 128 + k] : hg[tok * 128 + (k - 128)];
    sCat[i] = (__bf16)v;
  }
  __syncthreads();

  // L4: [64x256] @ [256x128] (relu) -> sY4
  {
    int col0 = wave * 16;
    bf16x16 Bf[8];
#pragma unroll
    for (int kt = 0; kt < 8; ++kt) Bf[kt] = load_bfrag_w(W4f, 8, wave, kt);
    float bias = b4[col0 + (lane & 15)];
#pragma unroll
    for (int rt = 0; rt < 4; ++rt) {
      f32x8 acc = splat8(bias);
#pragma unroll
      for (int kt = 0; kt < 8; ++kt)
        acc = WMMA_BF16(load_afrag(sCat, 256, rt * 16, kt * 32), Bf[kt], acc);
      store_c_lds_bf16(sY4, 128, rt * 16, col0, acc, true);
    }
  }
  __syncthreads();

  // L5: [64x128] @ [128x64] (relu) -> sY5   (8 waves: 4 coltiles x 2 rowtile-pairs)
  {
    int ct   = wave & 3;
    int col0 = ct * 16;
    int rt0  = (wave >> 2) * 2;
    bf16x16 Bf[4];
#pragma unroll
    for (int kt = 0; kt < 4; ++kt) Bf[kt] = load_bfrag_w(W5f, 4, ct, kt);
    float bias = b5[col0 + (lane & 15)];
#pragma unroll
    for (int rti = 0; rti < 2; ++rti) {
      f32x8 acc = splat8(bias);
#pragma unroll
      for (int kt = 0; kt < 4; ++kt)
        acc = WMMA_BF16(load_afrag(sY4, 128, (rt0 + rti) * 16, kt * 32), Bf[kt], acc);
      store_c_lds_bf16(sY5, 64, (rt0 + rti) * 16, col0, acc, true);
    }
  }
  __syncthreads();

  // L6: [64x64] @ [64x64] (relu) -> sY6 (f32)
  {
    int ct   = wave & 3;
    int col0 = ct * 16;
    int rt0  = (wave >> 2) * 2;
    bf16x16 Bf[2];
#pragma unroll
    for (int kt = 0; kt < 2; ++kt) Bf[kt] = load_bfrag_w(W6f, 2, ct, kt);
    float bias = b6[col0 + (lane & 15)];
#pragma unroll
    for (int rti = 0; rti < 2; ++rti) {
      f32x8 acc = splat8(bias);
#pragma unroll
      for (int kt = 0; kt < 2; ++kt)
        acc = WMMA_BF16(load_afrag(sY5, 64, (rt0 + rti) * 16, kt * 32), Bf[kt], acc);
      store_c_lds_f32(sY6, 64, (rt0 + rti) * 16, col0, acc, true);
    }
  }
  __syncthreads();

  // head: out = y6 @ Wo + bo  (64-dim dot per token)
  if (tid < 64) {
    float s = bo[0];
#pragma unroll 8
    for (int k = 0; k < 64; ++k) s += sY6[tid * 64 + k] * Wo[k];
    outg[tok0 + tid] = s;
  }
}

// ===========================================================================
extern "C" void kernel_launch(void* const* d_in, const int* in_sizes, int n_in,
                              void* d_out, int out_size, void* d_ws, size_t ws_size,
                              hipStream_t stream) {
  (void)n_in; (void)out_size; (void)ws_size;
  const float* states  = (const float*)d_in[0];
  const float* actions = (const float*)d_in[1];
  // d_in[2] = timesteps (unused by the reference computation)
  const float* W1 = (const float*)d_in[3];  const float* b1 = (const float*)d_in[4];
  const float* W2 = (const float*)d_in[5];  const float* b2 = (const float*)d_in[6];
  const float* W3 = (const float*)d_in[7];  const float* b3 = (const float*)d_in[8];
  const float* Wi = (const float*)d_in[9];
  const float* Wh = (const float*)d_in[10];
  const float* bl = (const float*)d_in[11];
  const float* W4 = (const float*)d_in[12]; const float* b4 = (const float*)d_in[13];
  const float* W5 = (const float*)d_in[14]; const float* b5 = (const float*)d_in[15];
  const float* W6 = (const float*)d_in[16]; const float* b6 = (const float*)d_in[17];
  const float* Wo = (const float*)d_in[18]; const float* bo = (const float*)d_in[19];

  const int BT = in_sizes[2];          // B*T (timesteps flat count)
  const int T  = 1000;                 // reference constant
  const int B  = BT / T;

  float* out      = (float*)d_out;     // [BT] reward, then [BT*128] lstm_out
  float* lstm_out = out + (size_t)BT;

  float* ws   = (float*)d_ws;
  float* x3g  = ws;                               // [BT*128] f32
  float* xwig = ws + (size_t)BT * 128;            // [BT*512] f32, LSTM-frag layout
  __bf16* wfrag = (__bf16*)(ws + (size_t)BT * (128 + 512));

  // bf16 B-fragment buffers (halves): [nct][nkt][32 lanes][16 halves]
  __bf16* W1f = wfrag;                 // 16ct x 2kt
  __bf16* W2f = W1f + 16 * 2 * 512;    //  8ct x 8kt
  __bf16* W3f = W2f + 8 * 8 * 512;     //  8ct x 4kt
  __bf16* Wif = W3f + 8 * 4 * 512;     // 32ct x 4kt
  __bf16* W4f = Wif + 32 * 4 * 512;    //  8ct x 8kt
  __bf16* W5f = W4f + 8 * 8 * 512;     //  4ct x 4kt
  __bf16* W6f = W5f + 4 * 4 * 512;     //  4ct x 2kt

  const int BIG = 1 << 30;
  swizzle_w_kernel<<<dim3(16, 2), 32, 0, stream>>>(W1, W1f, 256, 2, 37);
  swizzle_w_kernel<<<dim3(8, 8),  32, 0, stream>>>(W2, W2f, 128, 8, BIG);
  swizzle_w_kernel<<<dim3(8, 4),  32, 0, stream>>>(W3, W3f, 128, 4, BIG);
  swizzle_w_kernel<<<dim3(32, 4), 32, 0, stream>>>(Wi, Wif, 512, 4, BIG);
  swizzle_w_kernel<<<dim3(8, 8),  32, 0, stream>>>(W4, W4f, 128, 8, BIG);
  swizzle_w_kernel<<<dim3(4, 4),  32, 0, stream>>>(W5, W5f, 64, 4, BIG);
  swizzle_w_kernel<<<dim3(4, 2),  32, 0, stream>>>(W6, W6f, 64, 2, BIG);

  const size_t smemA = (size_t)(64 * 64 + 64 * 256 + 64 * 128 + 64 * 128) * 2;        // 72 KB
  const size_t smemB = (size_t)(128 * 512 + 2 * 16 * 128) * 2;                        // 136 KB
  const size_t smemC = (size_t)(64 * 256 + 64 * 128 + 64 * 64) * 2 + 64 * 64 * 4;     // 72 KB

  premlp_kernel<<<BT / 64, 256, smemA, stream>>>(states, actions, W1f, b1, W2f, b2, W3f, b3,
                                                 Wif, bl, x3g, xwig, T);
  lstm_kernel<<<B / 16, 256, smemB, stream>>>(Wh, xwig, lstm_out, T);
  postmlp_kernel<<<BT / 64, 256, smemC, stream>>>(x3g, lstm_out, W4f, b4, W5f, b5, W6f, b6,
                                                  Wo, bo, out);
}
